// AttentionRNN_76656576299207
// MI455X (gfx1250) — matrix operand order repaired and checked by reading, not compiled, not run on hardware
//
#include <hip/hip_runtime.h>
#include <hip/hip_bf16.h>

// ---------------------------------------------------------------------------
// AttentionRNN on gfx1250 (MI455X).
// B=4096, L=32, C=28, H=512, T=32.
// Dominant cost: per-step gates GEMM [4096 x 2048] = [4096 x 512] * [512 x 2048]
// v_wmma_f32_16x16x32_f16; each wave computes a 16x64 tile (4 accumulators),
// A (h) tile staged in LDS via CDNA5 async-to-LDS, B (w_hh) streamed from L2.
// ---------------------------------------------------------------------------

#define Bn   4096
#define Ln   32
#define Cn   28
#define Hn   512
#define Tn   32
#define G4H  2048   // 4*H

typedef _Float16 half8  __attribute__((ext_vector_type(8)));
typedef _Float16 half16 __attribute__((ext_vector_type(16)));
typedef float    float8 __attribute__((ext_vector_type(8)));
typedef int      v4i    __attribute__((ext_vector_type(4)));

#if defined(__has_builtin)
#if __has_builtin(__builtin_amdgcn_global_load_async_to_lds_b128)
#define HAVE_ASYNC_LDS 1
#endif
#if __has_builtin(__builtin_amdgcn_s_wait_asynccnt)
#define HAVE_WAIT_ASYNC 1
#endif
#endif

// 16-byte global -> LDS copy: async (ASYNCcnt-tracked) when available.
__device__ __forceinline__ void copy16_to_lds(const _Float16* g, _Float16* l) {
#if defined(HAVE_ASYNC_LDS)
    __builtin_amdgcn_global_load_async_to_lds_b128(
        (__attribute__((address_space(1))) v4i*)g,
        (__attribute__((address_space(3))) v4i*)l, 0, 0);
#else
    *(half8*)l = *(const half8*)g;
#endif
}

__device__ __forceinline__ void wait_async_copies() {
#if defined(HAVE_ASYNC_LDS)
#if defined(HAVE_WAIT_ASYNC)
    __builtin_amdgcn_s_wait_asynccnt(0);
#else
    asm volatile("s_wait_asynccnt 0" ::: "memory");
#endif
#endif
}

// ---------------------------------------------------------------------------
// Zero a float region (also used for f16 regions: 0 bits == 0.0h)
// ---------------------------------------------------------------------------
__global__ __launch_bounds__(256) void zero_f32(float* __restrict__ p, int n) {
    int i = blockIdx.x * 256 + threadIdx.x;
    if (i < n) p[i] = 0.0f;
}

// ---------------------------------------------------------------------------
// One-time weight prep: w_hh -> f16 [2048,512]; w_ih -> f16 padded [2048,32]
// (cols 28..31 zero so the context GEMM is a clean K=32 WMMA chunk);
// combined bias bc = b_ih + b_hh.
// ---------------------------------------------------------------------------
__global__ __launch_bounds__(256)
void prep_weights(const float* __restrict__ whh, const float* __restrict__ wih,
                  const float* __restrict__ bih, const float* __restrict__ bhh,
                  _Float16* __restrict__ whh16, _Float16* __restrict__ wih16,
                  float* __restrict__ bc) {
    int idx = blockIdx.x * 256 + threadIdx.x;
    if (idx < G4H * Hn) whh16[idx] = (_Float16)whh[idx];
    if (idx < G4H * 32) {
        int n = idx >> 5, k = idx & 31;
        wih16[idx] = (k < Cn) ? (_Float16)wih[n * Cn + k] : (_Float16)0.0f;
    }
    if (idx < G4H) bc[idx] = bih[idx] + bhh[idx];
}

// ---------------------------------------------------------------------------
// Preprocess: x = elu(conv1d(input,k=3,pad=1)*mask) + input   [B,L,C]
//             u_a = (x . conv_w + conv_b) * mask              [B,L]
// One thread per (b,l); conv1 weights staged in LDS.
// ---------------------------------------------------------------------------
__global__ __launch_bounds__(128)
void preprocess(const float* __restrict__ in, const float* __restrict__ mask,
                const float* __restrict__ c1w, const float* __restrict__ c1b,
                const float* __restrict__ cw, const float* __restrict__ cb,
                float* __restrict__ x, float* __restrict__ ua) {
    __shared__ float wS[Cn * Cn * 3];  // [o][i][d]
    __shared__ float cwS[Cn];
    for (int i = threadIdx.x; i < Cn * Cn * 3; i += 128) wS[i] = c1w[i];
    if (threadIdx.x < Cn) cwS[threadIdx.x] = cw[threadIdx.x];
    __syncthreads();

    int idx = blockIdx.x * 128 + threadIdx.x;           // flat (b,l)
    if (idx >= Bn * Ln) return;
    int b = idx >> 5, l = idx & 31;
    float m = mask[idx];
    float s = cb[0];
    const float* inrow = in + idx * Cn;
    for (int o = 0; o < Cn; ++o) {
        float acc = c1b[o];
        for (int d = 0; d < 3; ++d) {
            int lp = l + d - 1;
            if (lp < 0 || lp >= Ln) continue;
            const float* rp = in + (b * Ln + lp) * Cn;
            const float* wo = wS + o * (Cn * 3) + d;    // stride 3 over i
#pragma unroll
            for (int i2 = 0; i2 < Cn; ++i2) acc += rp[i2] * wo[i2 * 3];
        }
        float y = acc * m;
        float e = y > 0.0f ? y : (__expf(y) - 1.0f);    // ELU
        float xv = e + inrow[o];
        x[idx * Cn + o] = xv;
        s += xv * cwS[o];
    }
    ua[idx] = s * m;
}

// ---------------------------------------------------------------------------
// Per-step attention + context. One wave32 per batch row, 8 batches per block.
//   w_a = h.fc1_w + fc1_b ; attn = softmax(leaky_relu(u_a+w_a)+bias) over L=32
//   ctx[b,c] = sum_l attn[l]*x[b,l,c]; stored f16 padded to 32 (c 28..31 -> 0)
// Wave-synchronous: reductions and broadcast via __shfl (ds_bpermute/permlane).
// ---------------------------------------------------------------------------
__global__ __launch_bounds__(256)
void attention_ctx(const float* __restrict__ h32, const float* __restrict__ fc1w,
                   const float* __restrict__ fc1b, const float* __restrict__ ua,
                   const float* __restrict__ biasmat, const float* __restrict__ x,
                   _Float16* __restrict__ ctx16) {
    int b = blockIdx.x * 8 + (threadIdx.x >> 5);
    int l = threadIdx.x & 31;       // 0..31 == sequence position
    // dot(h[b], fc1_w) reduced across the wave
    float s = 0.0f;
    const float* hb = h32 + b * Hn;
    for (int k = l; k < Hn; k += 32) s += hb[k] * fc1w[k];
    for (int off = 16; off; off >>= 1) s += __shfl_xor(s, off, 32);
    float wa = s + fc1b[0];
    float z = ua[b * Ln + l] + wa;
    z = z > 0.0f ? z : 0.01f * z;                     // leaky_relu
    z += biasmat[b * Ln + l];
    float mx = z;
    for (int off = 16; off; off >>= 1) mx = fmaxf(mx, __shfl_xor(mx, off, 32));
    float e = __expf(z - mx);
    float se = e;
    for (int off = 16; off; off >>= 1) se += __shfl_xor(se, off, 32);
    float attn = e / se;
    // context: lane c accumulates over l; attn broadcast lane-by-lane
    float acc = 0.0f;
    const float* xb = x + (size_t)b * Ln * Cn;
    int cidx = (l < Cn) ? l : 0;
#pragma unroll
    for (int ll = 0; ll < Ln; ++ll)
        acc += __shfl(attn, ll, 32) * xb[ll * Cn + cidx];
    ctx16[b * 32 + l] = (l < Cn) ? (_Float16)acc : (_Float16)0.0f;
}

// ---------------------------------------------------------------------------
// Gates GEMM: G[4096,2048] = h16[4096,512] * whh16^T + ctx16[4096,32] * wih16^T
// Block = 4 waves; block tile 16(M) x 256(N); wave tile 16 x 64 (4 acc).
// A tile (16x512 f16 = 16 KB, contiguous rows) staged in LDS with the CDNA5
// async-to-LDS path (ASYNCcnt) when available, then read back as fragments
// via ds_load_b128. B streamed from global (L2-resident, 2 MB) with
// WGP-scope prefetch. WMMA fragment layouts used here (A/B k order is not the ISA table's):
//   A (16x32 f16): lane = hi*16+row; halves h -> K = hi*16+h (contiguous)
//                  (matches B order)
//   B (32x16 f16): lane = hi*16+col; halves h -> K = hi*16+h (contiguous)
//   C/D (16x16 f32): lane = hi*16+col; vgpr r -> M = hi*8+r
// ---------------------------------------------------------------------------
__global__ __launch_bounds__(128)
void gates_gemm(const _Float16* __restrict__ A,     // h16   [4096,512]
                const _Float16* __restrict__ W,     // whh16 [2048,512]
                const _Float16* __restrict__ Actx,  // ctx16 [4096,32]
                const _Float16* __restrict__ Wctx,  // wih16 [2048,32]
                float* __restrict__ G) {            // gates [4096,2048]
    __shared__ _Float16 As[16 * Hn];                // 16 KB
    __shared__ _Float16 Ac[16 * 32];                // 1 KB
    const int wave = threadIdx.x >> 5;
    const int lane = threadIdx.x & 31;
    const int tm = blockIdx.x >> 3;                 // 0..255  M tile
    const int tnb = blockIdx.x & 7;                 // 0..7    block N tile (256 wide)
    const int m0 = tm * 16;
    const int n0 = tnb * 256 + wave * 64;           // wave's 64-col slice
    const int rc = lane & 15;                       // row (A) / col (B,D)
    const int hi = lane >> 4;                       // K/M half select

    // ---- stage A tile (contiguous 8192 halves) + ctx tile (512 halves) ----
    {
        const _Float16* gsrc = A + (size_t)m0 * Hn;
        const _Float16* csrc = Actx + (size_t)m0 * 32;
        for (int i = threadIdx.x; i < 16 * Hn / 8; i += 128)
            copy16_to_lds(gsrc + i * 8, As + i * 8);
        if (threadIdx.x < 64)
            copy16_to_lds(csrc + threadIdx.x * 8, Ac + threadIdx.x * 8);
        wait_async_copies();
        __syncthreads();
    }

    float8 acc[4] = {};
    const _Float16* arow = As + rc * Hn;            // LDS
    const _Float16* brow = W + (size_t)(n0 + rc) * Hn;

    for (int k0 = 0; k0 < Hn; k0 += 32) {
        half8 alo = *(const half8*)(arow + k0 + hi * 16);         // ds_load_b128
        half8 ahi = *(const half8*)(arow + k0 + hi * 16 + 8);
        half16 a = __builtin_shufflevector(alo, ahi, 0, 1, 2, 3, 4, 5, 6, 7,
                                           8, 9, 10, 11, 12, 13, 14, 15);
        if (k0 + 32 < Hn) __builtin_prefetch(brow + k0 + 32, 0, 3);
#pragma unroll
        for (int sub = 0; sub < 4; ++sub) {
            half16 bfr = *(const half16*)(brow + (size_t)sub * 16 * Hn + k0 + hi * 16);
            acc[sub] = __builtin_amdgcn_wmma_f32_16x16x32_f16(
                false, a, false, bfr, (short)0, acc[sub], false, false);
        }
    }
    {   // context chunk, K = 32 (cols 28..31 zero-padded)
        half8 alo = *(const half8*)(Ac + rc * 32 + hi * 16);
        half8 ahi = *(const half8*)(Ac + rc * 32 + hi * 16 + 8);
        half16 a = __builtin_shufflevector(alo, ahi, 0, 1, 2, 3, 4, 5, 6, 7,
                                           8, 9, 10, 11, 12, 13, 14, 15);
#pragma unroll
        for (int sub = 0; sub < 4; ++sub) {
            half16 bfr = *(const half16*)(Wctx + (size_t)(n0 + sub * 16 + rc) * 32 + hi * 16);
            acc[sub] = __builtin_amdgcn_wmma_f32_16x16x32_f16(
                false, a, false, bfr, (short)0, acc[sub], false, false);
        }
    }
#pragma unroll
    for (int sub = 0; sub < 4; ++sub) {
        float* gbase = G + n0 + sub * 16 + rc;
#pragma unroll
        for (int r = 0; r < 8; ++r)
            gbase[(size_t)(m0 + hi * 8 + r) * G4H] = acc[sub][r];
    }
}

// ---------------------------------------------------------------------------
// Per-step LSTM cell + fused output projection (out[b,t] = h . conv2_w + b).
// One block of 512 threads per batch row; LDS tree reduction for the dot.
// ---------------------------------------------------------------------------
__global__ __launch_bounds__(512)
void lstm_update(const float* __restrict__ gates, const float* __restrict__ bc,
                 float* __restrict__ cst, float* __restrict__ h32,
                 _Float16* __restrict__ h16, const float* __restrict__ c2w,
                 const float* __restrict__ c2b, const float* __restrict__ mask,
                 float* __restrict__ out, int t) {
    __shared__ float red[Hn];
    int b = blockIdx.x, j = threadIdx.x;
    const float* g = gates + (size_t)b * G4H;
    float gi = g[j]            + bc[j];
    float gf = g[j + Hn]       + bc[j + Hn];
    float gg = g[j + 2 * Hn]   + bc[j + 2 * Hn];
    float go = g[j + 3 * Hn]   + bc[j + 3 * Hn];
    float ig = 1.0f / (1.0f + __expf(-gi));
    float fg = 1.0f / (1.0f + __expf(-gf));
    float gt = tanhf(gg);
    float og = 1.0f / (1.0f + __expf(-go));
    float cn = fg * cst[b * Hn + j] + ig * gt;
    cst[b * Hn + j] = cn;
    float h = og * tanhf(cn);
    h32[b * Hn + j] = h;
    h16[b * Hn + j] = (_Float16)h;
    red[j] = h * c2w[j];
    __syncthreads();
    for (int s = 256; s > 0; s >>= 1) {
        if (j < s) red[j] += red[j + s];
        __syncthreads();
    }
    if (j == 0) out[b * Tn + t] = (red[0] + c2b[0]) * mask[b * Ln + t];
}

// ---------------------------------------------------------------------------
// Host-side orchestration. All launches on `stream`; workspace carved by hand.
// ---------------------------------------------------------------------------
extern "C" void kernel_launch(void* const* d_in, const int* in_sizes, int n_in,
                              void* d_out, int out_size, void* d_ws, size_t ws_size,
                              hipStream_t stream) {
    const float* in      = (const float*)d_in[0];
    const float* mask    = (const float*)d_in[1];
    const float* biasmat = (const float*)d_in[2];
    const float* c1w     = (const float*)d_in[3];
    const float* c1b     = (const float*)d_in[4];
    const float* cw      = (const float*)d_in[5];
    const float* cb      = (const float*)d_in[6];
    const float* wih     = (const float*)d_in[7];
    const float* whh     = (const float*)d_in[8];
    const float* bih     = (const float*)d_in[9];
    const float* bhh     = (const float*)d_in[10];
    const float* fc1w    = (const float*)d_in[11];
    const float* fc1b    = (const float*)d_in[12];
    const float* c2w     = (const float*)d_in[13];
    const float* c2b     = (const float*)d_in[14];
    float* out = (float*)d_out;

    char* ws = (char*)d_ws;
    size_t off = 0;
    auto carve = [&](size_t bytes) {
        void* p = ws + off;
        off += (bytes + 255) & ~(size_t)255;
        return p;
    };
    float*    x      = (float*)   carve((size_t)Bn * Ln * Cn * 4);  // 14.7 MB
    float*    ua     = (float*)   carve((size_t)Bn * Ln * 4);       // 0.5 MB
    float*    gates  = (float*)   carve((size_t)Bn * G4H * 4);      // 32 MB
    float*    bc     = (float*)   carve((size_t)G4H * 4);
    // h32 / h16 / c contiguous so one zero kernel clears all three
    float*    zbase  = (float*)   carve((size_t)Bn * Hn * 4 +
                                        (size_t)Bn * Hn * 2 +
                                        (size_t)Bn * Hn * 4);       // 20 MB
    float*    h32    = zbase;
    _Float16* h16    = (_Float16*)((char*)zbase + (size_t)Bn * Hn * 4);
    float*    cst    = (float*)((char*)zbase + (size_t)Bn * Hn * 6);
    _Float16* whh16  = (_Float16*)carve((size_t)G4H * Hn * 2);      // 2 MB
    _Float16* wih16  = (_Float16*)carve((size_t)G4H * 32 * 2);
    _Float16* ctx16  = (_Float16*)carve((size_t)Bn * 32 * 2);

    // one-time prep (re-done every call: deterministic)
    prep_weights<<<(G4H * Hn + 255) / 256, 256, 0, stream>>>(
        whh, wih, bih, bhh, whh16, wih16, bc);
    preprocess<<<(Bn * Ln + 127) / 128, 128, 0, stream>>>(
        in, mask, c1w, c1b, cw, cb, x, ua);
    {   // zero h32 + h16 + c
        int nz = Bn * Hn + Bn * Hn / 2 + Bn * Hn;
        zero_f32<<<(nz + 255) / 256, 256, 0, stream>>>(zbase, nz);
    }

    // recurrent steps
    for (int t = 0; t < Tn; ++t) {
        attention_ctx<<<Bn / 8, 256, 0, stream>>>(
            h32, fc1w, fc1b, ua, biasmat, x, ctx16);
        // 256 M-tiles x 8 block-N-tiles (256 cols each) = 2048 blocks
        gates_gemm<<<2048, 128, 0, stream>>>(
            h16, whh16, ctx16, wih16, gates);
        lstm_update<<<Bn, Hn, 0, stream>>>(
            gates, bc, cst, h32, h16, c2w, c2b, mask, out, t);
    }
}
